// DotProductAttention_33122787787034
// MI455X (gfx1250) — matrix operand order repaired and checked by reading, not compiled, not run on hardware
//
#include <hip/hip_runtime.h>

typedef __attribute__((ext_vector_type(16))) _Float16 v16h;
typedef __attribute__((ext_vector_type(8)))  _Float16 v8h;
typedef __attribute__((ext_vector_type(2)))  _Float16 h2;
typedef __attribute__((ext_vector_type(8)))  float    v8f;

#define B_     32
#define T_     2048
#define D_     64
#define QTILE  128   // query rows per block (8 waves x 16 rows)
#define CHUNK  32    // key rows per LDS tile (= WMMA K for the P*V matmul)
#define NWAVES 8

#ifndef __has_builtin
#define __has_builtin(x) 0
#endif

// ---------------- gfx1250 async global->LDS path (guarded) ----------------
#if __has_builtin(__builtin_amdgcn_global_load_async_to_lds_b128)
#define HAVE_ASYNC_LDS 1
typedef int v4i __attribute__((vector_size(16)));
typedef __attribute__((address_space(1))) v4i gas_v4i;   // global int4
typedef __attribute__((address_space(3))) v4i las_v4i;   // LDS int4
#define ASYNC_CP16(gp, lp) \
    __builtin_amdgcn_global_load_async_to_lds_b128((gas_v4i*)(gp), (las_v4i*)(lp), 0, 0)
#else
#define HAVE_ASYNC_LDS 0
#endif

#if __has_builtin(__builtin_amdgcn_s_wait_asynccnt)
#define WAIT_ASYNC(n) __builtin_amdgcn_s_wait_asynccnt(n)
#else
#define WAIT_ASYNC(n) asm volatile("s_wait_asynccnt %0" ::"i"(n) : "memory")
#endif

// ---------------- DPP16 ROW_XMASK butterfly (16-lane row reduce) ----------
#if __has_builtin(__builtin_amdgcn_mov_dpp)
template <int XMASK>
__device__ __forceinline__ float dpp_xor16(float x) {
    return __int_as_float(__builtin_amdgcn_mov_dpp(
        __float_as_int(x), 0x160 + XMASK, 0xf, 0xf, true));
}
__device__ __forceinline__ float rowmax16(float v) {
    v = fmaxf(v, dpp_xor16<1>(v));
    v = fmaxf(v, dpp_xor16<2>(v));
    v = fmaxf(v, dpp_xor16<4>(v));
    v = fmaxf(v, dpp_xor16<8>(v));
    return v;
}
__device__ __forceinline__ float rowsum16(float v) {
    v += dpp_xor16<1>(v);
    v += dpp_xor16<2>(v);
    v += dpp_xor16<4>(v);
    v += dpp_xor16<8>(v);
    return v;
}
#else
__device__ __forceinline__ float rowmax16(float v) {
    v = fmaxf(v, __shfl_xor(v, 1, 32));
    v = fmaxf(v, __shfl_xor(v, 2, 32));
    v = fmaxf(v, __shfl_xor(v, 4, 32));
    v = fmaxf(v, __shfl_xor(v, 8, 32));
    return v;
}
__device__ __forceinline__ float rowsum16(float v) {
    v += __shfl_xor(v, 1, 32);
    v += __shfl_xor(v, 2, 32);
    v += __shfl_xor(v, 4, 32);
    v += __shfl_xor(v, 8, 32);
    return v;
}
#endif

#if __has_builtin(__builtin_amdgcn_exp2f)
#define EXP2F(x) __builtin_amdgcn_exp2f(x)
#else
#define EXP2F(x) exp2f(x)
#endif

// packed f32x2 -> f16x2 convert (v_cvt_pk_rtz_f16_f32); bit-cast __fp16x2 -> _Float16x2
#if __has_builtin(__builtin_amdgcn_cvt_pkrtz)
__device__ __forceinline__ h2 CVT_PK(float a, float b) {
    return __builtin_bit_cast(h2, __builtin_amdgcn_cvt_pkrtz(a, b));
}
#else
__device__ __forceinline__ h2 CVT_PK(float a, float b) {
    h2 r; r[0] = (_Float16)a; r[1] = (_Float16)b; return r;
}
#endif

__global__ __launch_bounds__(256) void fa_causal_wmma(
    const float* __restrict__ Q,
    const float* __restrict__ K,
    const float* __restrict__ V,
    float* __restrict__ O)
{
    // raw fp32 double buffers fed by the async engine, f16 compute tiles, P staging
    __shared__ __align__(128) float     sRawK[2][CHUNK][D_];      // 2 x 8 KB
    __shared__ __align__(128) float     sRawV[2][CHUNK][D_];      // 2 x 8 KB
    __shared__ __align__(128) _Float16  sK [CHUNK][D_];           // 4 KB
    __shared__ __align__(128) _Float16  sVt[D_][CHUNK];           // 4 KB
    __shared__ __align__(128) _Float16  sP [NWAVES][16][CHUNK];   // 8 KB

    const int tid   = threadIdx.x;
    const int lane  = tid & 31;
    const int wave  = tid >> 5;
    const int khalf = lane >> 4;   // lane-half select per ISA layouts
    const int lm    = lane & 15;

    const int blocksPerBatch = T_ / QTILE;
    const int batch  = blockIdx.x / blocksPerBatch;
    const int qtile0 = (blockIdx.x % blocksPerBatch) * QTILE;
    const int qrow0  = qtile0 + wave * 16;

    const float LOG2E  = 1.44269504088896f;
    const float qscale = 0.125f * LOG2E;   // 1/sqrt(64), log2-domain softmax

    const float* Qb = Q + (size_t)batch * T_ * D_;
    const float* Kb = K + (size_t)batch * T_ * D_;
    const float* Vb = V + (size_t)batch * T_ * D_;
    float*       Ob = O + (size_t)batch * T_ * D_;

    // per-thread staging slice (owns both async destination and convert reads)
    const int sr  = tid >> 3;        // key row 0..31
    const int sc8 = (tid & 7) * 8;   // dim base 0..56

    // ---- Q tile (16 x 64) -> two WMMA A operands, scale folded in ----
    // A layout: lane holds row M = lane%16; element 2v+e is K = 2v + 16*khalf + e
    v16h qa[2];
    {
        const float* qrow = Qb + (size_t)(qrow0 + lm) * D_;
        #pragma unroll
        for (int c = 0; c < 2; ++c) {
            #pragma unroll
            for (int v = 0; v < 8; ++v) {
                int kb = 2 * v + 16 * khalf;
                qa[c][2 * v + 0] = (_Float16)(qrow[c * 32 + kb + 0] * qscale);
                qa[c][2 * v + 1] = (_Float16)(qrow[c * 32 + kb + 1] * qscale);
            }
        }
    }

    // online-softmax state: row j of this lane-half (global row = qrow0 + j + 8*khalf)
    float mrow[8], lrow[8];
    int   climit[8];   // causal threshold per row: mask e0 if jkey > climit[j]
    v8f acc[4];
    #pragma unroll
    for (int j = 0; j < 8; ++j) {
        mrow[j]   = -1e30f;
        lrow[j]   = 0.0f;
        climit[j] = qrow0 + j + 8 * khalf - lm;   // grow - lm (loop-invariant)
    }
    #pragma unroll
    for (int t = 0; t < 4; ++t)
        #pragma unroll
        for (int j = 0; j < 8; ++j) acc[t][j] = 0.0f;

    const int jend = qtile0 + QTILE;  // block's causal frontier

    // ---- stage helper (async copy when available, sync fallback otherwise) ----
    auto stage = [&](int jkey, int buf) {
        const float* kg = Kb + (size_t)(jkey + sr) * D_ + sc8;
        const float* vg = Vb + (size_t)(jkey + sr) * D_ + sc8;
#if HAVE_ASYNC_LDS
        ASYNC_CP16(kg,     &sRawK[buf][sr][sc8]);
        ASYNC_CP16(kg + 4, &sRawK[buf][sr][sc8 + 4]);
        ASYNC_CP16(vg,     &sRawV[buf][sr][sc8]);
        ASYNC_CP16(vg + 4, &sRawV[buf][sr][sc8 + 4]);
#else
        *(float4*)&sRawK[buf][sr][sc8]     = *(const float4*)kg;
        *(float4*)&sRawK[buf][sr][sc8 + 4] = *(const float4*)(kg + 4);
        *(float4*)&sRawV[buf][sr][sc8]     = *(const float4*)vg;
        *(float4*)&sRawV[buf][sr][sc8 + 4] = *(const float4*)(vg + 4);
#endif
    };

    stage(0, 0);  // prologue: chunk 0 in flight

    int it = 0;
    for (int jkey = 0; jkey < jend; jkey += CHUNK, ++it) {
        const int  buf     = it & 1;
        const bool hasNext = (jkey + CHUNK) < jend;

        if (hasNext) {
            stage(jkey + CHUNK, buf ^ 1);   // next chunk streams while we compute
            WAIT_ASYNC(4);                  // current chunk's 4 copies complete
        } else {
            WAIT_ASYNC(0);
        }
        __syncthreads();                    // raw[buf] visible; prior compute done

        // ---- convert raw fp32 -> f16 tiles (K row-major, V transposed) ----
        {
            const float4* kr = (const float4*)&sRawK[buf][sr][sc8];
            float4 k0 = kr[0], k1 = kr[1];
            union { v8h v; h2 h[4]; } kh;
            kh.h[0] = CVT_PK(k0.x, k0.y);
            kh.h[1] = CVT_PK(k0.z, k0.w);
            kh.h[2] = CVT_PK(k1.x, k1.y);
            kh.h[3] = CVT_PK(k1.z, k1.w);
            *(v8h*)&sK[sr][sc8] = kh.v;

            const float4* vr = (const float4*)&sRawV[buf][sr][sc8];
            float4 v0 = vr[0], v1 = vr[1];
            float ve[8] = {v0.x, v0.y, v0.z, v0.w, v1.x, v1.y, v1.z, v1.w};
            #pragma unroll
            for (int e = 0; e < 8; ++e) sVt[sc8 + e][sr] = (_Float16)ve[e];
        }
        __syncthreads();                    // f16 tiles ready for all waves

        if (jkey <= qrow0 + 15) {  // wave-uniform causal skip; EXEC all-1s for WMMA
            // ---- S = Q K^T : two 16x16 score tiles over this 32-key chunk ----
            v8f s0 = {}, s1 = {};
            #pragma unroll
            for (int c = 0; c < 2; ++c) {
                v16h kb0 = *(const v16h*)&sK[ 0 + lm][c * 32 + 16 * khalf];
                v16h kb1 = *(const v16h*)&sK[16 + lm][c * 32 + 16 * khalf];
                s0 = __builtin_amdgcn_wmma_f32_16x16x32_f16(false, qa[c], false, kb0,
                                                            (short)0, s0, false, false);
                s1 = __builtin_amdgcn_wmma_f32_16x16x32_f16(false, qa[c], false, kb1,
                                                            (short)0, s1, false, false);
            }

            // ---- causal mask: only the frontier chunk pays for it ----
            if (jkey + CHUNK - 1 > qrow0) {       // wave-uniform branch
                #pragma unroll
                for (int j = 0; j < 8; ++j) {
                    if (jkey      > climit[j]) s0[j] = -1e30f;
                    if (jkey + 16 > climit[j]) s1[j] = -1e30f;
                }
            }

            // ---- online softmax (DPP16 row reductions, log2 domain) ----
            #pragma unroll
            for (int j = 0; j < 8; ++j) {
                int rowm = j + 8 * khalf;
                float e0 = s0[j], e1 = s1[j];
                float mnew = fmaxf(mrow[j], rowmax16(fmaxf(e0, e1)));
                float corr = EXP2F(mrow[j] - mnew);
                mrow[j] = mnew;
                float p0 = EXP2F(e0 - mnew);
                float p1 = EXP2F(e1 - mnew);
                lrow[j] = lrow[j] * corr + rowsum16(p0 + p1);
                #pragma unroll
                for (int t = 0; t < 4; ++t) acc[t][j] *= corr;
                // stage P in f16, C-layout -> row-major (per-wave buffer)
                sP[wave][rowm][ 0 + lm] = (_Float16)p0;
                sP[wave][rowm][16 + lm] = (_Float16)p1;
            }
            asm volatile("s_wait_dscnt 0" ::: "memory");  // same-wave LDS RAW

            // ---- reload P as a WMMA A operand (16 x 32) ----
            v8h plo = *(const v8h*)&sP[wave][lm][16 * khalf + 0];
            v8h phi = *(const v8h*)&sP[wave][lm][16 * khalf + 8];
            v16h pa;
            #pragma unroll
            for (int i = 0; i < 8; ++i) { pa[i] = plo[i]; pa[8 + i] = phi[i]; }

            // ---- acc += P * V : four independent 16x16 output tiles (D = 64) ----
            #pragma unroll
            for (int t = 0; t < 4; ++t) {
                v16h vb = *(const v16h*)&sVt[t * 16 + lm][16 * khalf];
                acc[t] = __builtin_amdgcn_wmma_f32_16x16x32_f16(false, pa, false, vb,
                                                                (short)0, acc[t], false, false);
            }
        }
        __syncthreads();                    // tiles consumed before next convert
    }

    // ---- epilogue: O = acc / l ----
    #pragma unroll
    for (int j = 0; j < 8; ++j) {
        int rowm = j + 8 * khalf;
        float inv = 1.0f / lrow[j];
        #pragma unroll
        for (int t = 0; t < 4; ++t)
            Ob[(size_t)(qrow0 + rowm) * D_ + t * 16 + lm] = acc[t][j] * inv;
    }
}

extern "C" void kernel_launch(void* const* d_in, const int* in_sizes, int n_in,
                              void* d_out, int out_size, void* d_ws, size_t ws_size,
                              hipStream_t stream) {
    const float* Q = (const float*)d_in[0];
    const float* K = (const float*)d_in[1];
    const float* V = (const float*)d_in[2];
    float*       O = (float*)d_out;
    dim3 grid(B_ * (T_ / QTILE));  // 512 blocks
    dim3 block(256);               // 8 wave32 waves
    fa_causal_wmma<<<grid, block, 0, stream>>>(Q, K, V, O);
}
